// COMI_42090679501337
// MI455X (gfx1250) — matrix operand order repaired and checked by reading, not compiled, not run on hardware
//
#include <hip/hip_runtime.h>
#include <cstdint>
#include <cstddef>

#define B_  1024
#define S_  50
#define K_  5
#define D_  128
#define NI_ 100000
#define C_  640   // K_*D_

typedef __attribute__((ext_vector_type(16))) __bf16        v16bf;
typedef __attribute__((ext_vector_type(8)))  float         v8f;
typedef __attribute__((ext_vector_type(4)))  unsigned int  u32x4;
typedef __attribute__((ext_vector_type(4)))  float         f32x4;

union Frag { v16bf f; u32x4 q[2]; };

__device__ __forceinline__ unsigned short f32_to_bf16_rne(float x) {
    unsigned int u = __float_as_uint(x);
    unsigned int r = (u + 0x7FFFu + ((u >> 16) & 1u)) >> 16;
    return (unsigned short)r;
}

__device__ __forceinline__ float waveRedSum32(float v) {
#pragma unroll
    for (int m = 16; m >= 1; m >>= 1) v += __shfl_xor(v, m, 32);
    return v;
}

__device__ __forceinline__ v8f zero8() {
    v8f z;
#pragma unroll
    for (int i = 0; i < 8; ++i) z[i] = 0.0f;
    return z;
}

// ---------------------------------------------------------------- conversion
__global__ void conv_f32_bf16(const float* __restrict__ src,
                              unsigned short* __restrict__ dst, int n) {
    int i = blockIdx.x * blockDim.x + threadIdx.x;
    if (i < n) dst[i] = f32_to_bf16_rne(src[i]);
}

// ---------------------------------------------------------------- hat GEMMs
// grid (B/16, S), block 256.  hat[(b*S+s)*640 + col] = sum_d seq[b,s,d]*w[s,col,d]
__global__ void hat_kernel(const int* __restrict__ item_seq,
                           const unsigned short* __restrict__ emb16,
                           const unsigned short* __restrict__ w16,
                           float* __restrict__ hat) {
    __shared__ u32x4 aT4[256];        // 16 rows x 128 bf16
    __shared__ int   idxs[16];

    const int b0 = blockIdx.x * 16;
    const int s  = blockIdx.y;
    const int t  = threadIdx.x;

    if (t < 16) idxs[t] = item_seq[(b0 + t) * S_ + s];
    __syncthreads();
    {   // gather 16 rows of item_emb (bf16) into LDS; 256 threads x 16B
        const int r = t >> 4, c = t & 15;
        aT4[t] = ((const u32x4*)emb16)[(size_t)idxs[r] * 16 + c];
    }
    __syncthreads();

    const int wave = t >> 5, lane = t & 31;
    const int lm = lane & 15, half = lane >> 4;
    const int klaA = half * 16;   // A: contiguous 16-K run
    const int klaB = half * 16;   // B: contiguous 16-K run

    const unsigned short* aRow = ((const unsigned short*)aT4) + lm * 128;

    v8f acc[5];
#pragma unroll
    for (int i = 0; i < 5; ++i) acc[i] = zero8();

#pragma unroll
    for (int kk = 0; kk < 4; ++kk) {
        Frag A;
        const u32x4* ap = (const u32x4*)(aRow + kk * 32 + klaA);
        A.q[0] = ap[0];
        A.q[1] = ap[1];               // +8 elements = +1 u32x4
#pragma unroll
        for (int tt = 0; tt < 5; ++tt) {
            const int n0 = (wave * 5 + tt) * 16;
            const unsigned short* bp =
                w16 + ((size_t)s * C_ + n0 + lm) * 128 + kk * 32 + klaB;
            Frag Bf;
            Bf.q[0] = ((const u32x4*)bp)[0];
            Bf.q[1] = ((const u32x4*)bp)[1];
            acc[tt] = __builtin_amdgcn_wmma_f32_16x16x32_bf16(
                false, A.f, false, Bf.f, (short)0, acc[tt], false, false);
        }
    }

#pragma unroll
    for (int tt = 0; tt < 5; ++tt) {
        const int n = (wave * 5 + tt) * 16 + lm;
#pragma unroll
        for (int r = 0; r < 8; ++r) {
            const int m = r + half * 8;
            hat[((size_t)(b0 + m) * S_ + s) * C_ + n] = acc[tt][r];
        }
    }
}

// ---------------------------------------------------------------- routing
// grid B/8, block 256 (one wave per batch row)
__global__ void routing_kernel(const float* __restrict__ hat,
                               const float* __restrict__ mask,
                               const int* __restrict__ item,
                               const float* __restrict__ item_emb,
                               float* __restrict__ out_cap,
                               unsigned short* __restrict__ best16) {
    __shared__ float logitsL[8][K_ * S_];
    __shared__ float wsoftL[8][K_ * S_];
    __shared__ float maskL[8][S_];

    const int t = threadIdx.x;
    const int w = t >> 5, lane = t & 31;
    const int b = blockIdx.x * 8 + w;

    for (int i = lane; i < K_ * S_; i += 32) logitsL[w][i] = 0.0f;
    for (int s = lane; s < S_; s += 32)      maskL[w][s] = mask[(size_t)b * S_ + s];
    __syncthreads();

    f32x4 cap[K_];
    const float* hb = hat + (size_t)b * S_ * C_;

    for (int it = 0; it < 3; ++it) {
        // softmax over K per position s, masked
        for (int s = lane; s < S_; s += 32) {
            float l[K_], mx = -3.4e38f;
#pragma unroll
            for (int k = 0; k < K_; ++k) { l[k] = logitsL[w][k * S_ + s]; mx = fmaxf(mx, l[k]); }
            float sum = 0.0f;
#pragma unroll
            for (int k = 0; k < K_; ++k) { l[k] = __expf(l[k] - mx); sum += l[k]; }
            const float inv = 1.0f / sum;
            const float mk = maskL[w][s];
#pragma unroll
            for (int k = 0; k < K_; ++k)
                wsoftL[w][k * S_ + s] = (mk == 0.0f) ? 0.0f : l[k] * inv;
        }
        __syncthreads();

        // cap[k] = squash(sum_s w_soft * hat)
#pragma unroll
        for (int k = 0; k < K_; ++k) {
            f32x4 a; a.x = a.y = a.z = a.w = 0.0f;
            for (int s = 0; s < S_; ++s) {
                const float wv = wsoftL[w][k * S_ + s];
                const f32x4 h = *(const f32x4*)(hb + (size_t)s * C_ + k * D_ + lane * 4);
                a.x += wv * h.x; a.y += wv * h.y; a.z += wv * h.z; a.w += wv * h.w;
            }
            const float n2 = waveRedSum32(a.x * a.x + a.y * a.y + a.z * a.z + a.w * a.w);
            const float sc = (n2 / (1.0f + n2)) * rsqrtf(n2 + 1e-9f);
            a.x *= sc; a.y *= sc; a.z *= sc; a.w *= sc;
            cap[k] = a;
        }

        if (it < 2) {
#pragma unroll
            for (int k = 0; k < K_; ++k) {
                for (int s = 0; s < S_; ++s) {
                    const f32x4 h = *(const f32x4*)(hb + (size_t)s * C_ + k * D_ + lane * 4);
                    const float p = waveRedSum32(h.x * cap[k].x + h.y * cap[k].y +
                                                 h.z * cap[k].z + h.w * cap[k].w);
                    if (lane == 0) logitsL[w][k * S_ + s] += p;
                }
            }
        }
        __syncthreads();
    }

    // write multi_interest_emb
#pragma unroll
    for (int k = 0; k < K_; ++k)
        *(f32x4*)(out_cap + ((size_t)b * K_ + k) * D_ + lane * 4) = cap[k];

    // cos with target item, argmax, gather best interest (bf16 for scoring GEMM)
    const int tg = item[b];
    const f32x4 e = *(const f32x4*)(item_emb + (size_t)tg * D_ + lane * 4);
    float bestv = -3.4e38f; int kb = 0;
#pragma unroll
    for (int k = 0; k < K_; ++k) {
        const float c = waveRedSum32(cap[k].x * e.x + cap[k].y * e.y +
                                     cap[k].z * e.z + cap[k].w * e.w);
        if (c > bestv) { bestv = c; kb = k; }
    }
    f32x4 bb = cap[0];
#pragma unroll
    for (int k = 1; k < K_; ++k) if (kb == k) bb = cap[k];
    unsigned short* bp = best16 + (size_t)b * D_ + lane * 4;
    bp[0] = f32_to_bf16_rne(bb.x); bp[1] = f32_to_bf16_rne(bb.y);
    bp[2] = f32_to_bf16_rne(bb.z); bp[3] = f32_to_bf16_rne(bb.w);
}

// ---------------------------------------------------------------- scores+loss
// grid B/16, block 256; streams all NI_ items in 16-col WMMA tiles,
// online logsumexp per row.
__global__ void score_kernel(const unsigned short* __restrict__ best16,
                             const unsigned short* __restrict__ emb16,
                             const int* __restrict__ item,
                             float* __restrict__ lossPart) {
    __shared__ u32x4 aT4[256];
    __shared__ int   tgt[16];
    __shared__ float tgtScore[16];
    __shared__ float wm[8][16], wsum[8][16];

    const int b0 = blockIdx.x * 16;
    const int t  = threadIdx.x;

    {   const int r = t >> 4, c = t & 15;
        aT4[t] = ((const u32x4*)best16)[(size_t)(b0 + r) * 16 + c]; }
    if (t < 16) { tgt[t] = item[b0 + t]; tgtScore[t] = 0.0f; }
    __syncthreads();

    const int wave = t >> 5, lane = t & 31;
    const int lm = lane & 15, half = lane >> 4;
    const int klaA = half * 16, klaB = half * 16;

    // hoist the 4 A fragments (fixed for the whole N sweep)
    Frag Af[4];
    const unsigned short* aRow = ((const unsigned short*)aT4) + lm * 128;
#pragma unroll
    for (int kk = 0; kk < 4; ++kk) {
        const u32x4* ap = (const u32x4*)(aRow + kk * 32 + klaA);
        Af[kk].q[0] = ap[0];
        Af[kk].q[1] = ap[1];
    }

    int trow[8];
#pragma unroll
    for (int r = 0; r < 8; ++r) trow[r] = tgt[r + half * 8];

    float rm[8], rs[8];
#pragma unroll
    for (int r = 0; r < 8; ++r) { rm[r] = -3.4e38f; rs[r] = 0.0f; }

    for (int chunk = wave; chunk < NI_ / 16; chunk += 8) {
        const int n0 = chunk * 16;
        const int col = n0 + lm;
        if (n0 + 128 + lm < NI_)
            __builtin_prefetch(emb16 + (size_t)(n0 + 128 + lm) * 128, 0, 0);

        v8f acc = zero8();
        const unsigned short* brow = emb16 + (size_t)col * 128 + klaB;
#pragma unroll
        for (int kk = 0; kk < 4; ++kk) {
            Frag Bf;
            Bf.q[0] = ((const u32x4*)(brow + kk * 32))[0];
            Bf.q[1] = ((const u32x4*)(brow + kk * 32))[1];
            acc = __builtin_amdgcn_wmma_f32_16x16x32_bf16(
                false, Af[kk].f, false, Bf.f, (short)0, acc, false, false);
        }
#pragma unroll
        for (int r = 0; r < 8; ++r) {
            const float x = acc[r];
            if (col == trow[r]) tgtScore[r + half * 8] = x;
            const float nm = fmaxf(rm[r], x);
            rs[r] = rs[r] * __expf(rm[r] - nm) + __expf(x - nm);
            rm[r] = nm;
        }
    }

    // combine across the 16 lanes that share each row (stay within lane-halves)
#pragma unroll
    for (int r = 0; r < 8; ++r) {
#pragma unroll
        for (int o = 1; o < 16; o <<= 1) {
            const float om = __shfl_xor(rm[r], o, 16);
            const float os = __shfl_xor(rs[r], o, 16);
            const float nm = fmaxf(rm[r], om);
            rs[r] = rs[r] * __expf(rm[r] - nm) + os * __expf(om - nm);
            rm[r] = nm;
        }
    }
    if (lm == 0) {
#pragma unroll
        for (int r = 0; r < 8; ++r) {
            wm[wave][r + half * 8] = rm[r];
            wsum[wave][r + half * 8] = rs[r];
        }
    }
    __syncthreads();

    if (t < 16) {   // deterministic fixed-order combine across 8 waves
        float m = -3.4e38f, s = 0.0f;
#pragma unroll
        for (int w = 0; w < 8; ++w) {
            const float nm = fmaxf(m, wm[w][t]);
            s = s * __expf(m - nm) + wsum[w][t] * __expf(wm[w][t] - nm);
            m = nm;
        }
        lossPart[b0 + t] = -(tgtScore[t] - m - __logf(s)) * (1.0f / (float)B_);
    }
}

// ---------------------------------------------------------------- final sum
__global__ void loss_reduce(const float* __restrict__ lossPart,
                            float* __restrict__ out_loss) {
    __shared__ float red[256];
    const int t = threadIdx.x;
    red[t] = lossPart[t] + lossPart[t + 256] + lossPart[t + 512] + lossPart[t + 768];
    __syncthreads();
#pragma unroll
    for (int o = 128; o >= 1; o >>= 1) {
        if (t < o) red[t] += red[t + o];
        __syncthreads();
    }
    if (t == 0) *out_loss = red[0];
}

// ---------------------------------------------------------------- launcher
extern "C" void kernel_launch(void* const* d_in, const int* in_sizes, int n_in,
                              void* d_out, int out_size, void* d_ws, size_t ws_size,
                              hipStream_t stream) {
    const int*   item_seq = (const int*)d_in[0];
    const float* mask     = (const float*)d_in[1];
    const int*   item     = (const int*)d_in[2];
    const float* item_emb = (const float*)d_in[3];
    const float* w        = (const float*)d_in[4];
    float*       out      = (float*)d_out;

    char* ws = (char*)d_ws;
    unsigned short* emb16  = (unsigned short*)(ws);                       // 25,600,000 B
    unsigned short* w16    = (unsigned short*)(ws + 25600000);            //  8,192,000 B
    float*          hat    = (float*)(ws + 33792000);                     // 131,072,000 B
    unsigned short* best16 = (unsigned short*)(ws + 164864000);           //    262,144 B
    float*          lossP  = (float*)(ws + 165126144);                    //      4,096 B

    conv_f32_bf16<<<(NI_ * D_ + 255) / 256, 256, 0, stream>>>(item_emb, emb16, NI_ * D_);
    conv_f32_bf16<<<(S_ * C_ * D_ + 255) / 256, 256, 0, stream>>>(w, w16, S_ * C_ * D_);
    hat_kernel<<<dim3(B_ / 16, S_), 256, 0, stream>>>(item_seq, emb16, w16, hat);
    routing_kernel<<<B_ / 8, 256, 0, stream>>>(hat, mask, item, item_emb, out, best16);
    score_kernel<<<B_ / 16, 256, 0, stream>>>(best16, emb16, item, lossP);
    loss_reduce<<<1, 256, 0, stream>>>(lossP, out + (size_t)B_ * K_ * D_);
}
